// InterpretableMultiHeadAttention_14946486190115
// MI455X (gfx1250) — matrix operand order repaired and checked by reading, not compiled, not run on hardware
//
#include <hip/hip_runtime.h>
#include <hip/hip_bf16.h>

typedef __attribute__((ext_vector_type(16))) _Float16 v16h;
typedef __attribute__((ext_vector_type(8)))  _Float16 v8h;
typedef __attribute__((ext_vector_type(8)))  float    v8f;
typedef __attribute__((ext_vector_type(4)))  int      v4i;

#define B_   4
#define S_   2048
#define H_   16
#define DK_  64
#define NO_  1024   // N_OUT

#if defined(__AMDGCN__) && __has_builtin(__builtin_amdgcn_global_load_async_to_lds_b128)
#define HAVE_ASYNC_LDS 1
typedef __attribute__((address_space(1))) v4i gv4i;  // global (prints __device__)
typedef __attribute__((address_space(3))) v4i lv4i;  // LDS   (prints __shared__)
#else
#define HAVE_ASYNC_LDS 0
#endif

// ---------------------------------------------------------------------------
// helpers
// ---------------------------------------------------------------------------
__device__ __forceinline__ v8f wmma16x16x32(v16h a, v16h b, v8f c) {
  // (neg_a, A, neg_b, B, c_mod, C, reuse_a, reuse_b)
  return __builtin_amdgcn_wmma_f32_16x16x32_f16(false, a, false, b, (short)0, c,
                                                false, false);
}

__device__ __forceinline__ v16h cat16(v8h lo, v8h hi) {
  return __builtin_shufflevector(lo, hi, 0, 1, 2, 3, 4, 5, 6, 7, 8, 9, 10, 11,
                                 12, 13, 14, 15);
}

__device__ __forceinline__ v8h ldg8h(const _Float16* p) {
  return *(const v8h*)p;
}

// A-fragment (16x32 f16): lane m = row, halves at k0+half*8(+0..7) and
// k0+16+half*8(+0..7).   B-fragment (32x16 f16): lane n = col, halves at
// k0+half*8(+0..7) and k0+16+half*8(+0..7), where memory holds B^T rows (n-major, contiguous k).
__device__ __forceinline__ v16h afrag(const _Float16* row, int k0, int half) {
  return cat16(ldg8h(row + k0 + half * 8), ldg8h(row + k0 + 16 + half * 8));
}
__device__ __forceinline__ v16h bfrag(const _Float16* row, int k0, int half) {
  return cat16(ldg8h(row + k0 + half * 8), ldg8h(row + k0 + 16 + half * 8));
}

// Stage one contiguous 4KB tile (32 keys x 64 halves) global -> LDS.
// Async DMA path on gfx1250; synchronous vector-copy fallback elsewhere.
__device__ __forceinline__ void stage_tile_4k(const _Float16* __restrict__ src,
                                              _Float16* dst, int lane) {
#if HAVE_ASYNC_LDS
#pragma unroll
  for (int i = 0; i < 8; ++i) {
    // generic->AS1 is bit-identical; generic->AS3 keeps the low 32 bits,
    // which per the flat-addressing rules are the LDS byte offset.
    gv4i* g = (gv4i*)(unsigned long long)(const void*)(src + i * 256 + lane * 8);
    lv4i* l = (lv4i*)(unsigned)(unsigned long long)(const void*)
                  (dst + i * 256 + lane * 8);
    __builtin_amdgcn_global_load_async_to_lds_b128(g, l, 0, 0);
  }
#else
#pragma unroll
  for (int i = 0; i < 8; ++i)
    *(v8h*)(dst + i * 256 + lane * 8) = ldg8h(src + i * 256 + lane * 8);
#endif
}

__device__ __forceinline__ void wait_async_le8() {
#if HAVE_ASYNC_LDS
  asm volatile("s_wait_asynccnt 0x8" ::: "memory");
#endif
}
__device__ __forceinline__ void wait_async_le0() {
#if HAVE_ASYNC_LDS
  asm volatile("s_wait_asynccnt 0x0" ::: "memory");
#endif
}

// ---------------------------------------------------------------------------
// Kernel 0: elementwise fp32 -> f16 (8 elements/thread); n % 2048 == 0
// ---------------------------------------------------------------------------
__global__ __launch_bounds__(256)
void cvt_f16_kernel(const float* __restrict__ in, _Float16* __restrict__ out) {
  const size_t i = ((size_t)blockIdx.x * 256 + threadIdx.x) * 8;
  v8h r;
#pragma unroll
  for (int j = 0; j < 8; ++j) r[j] = (_Float16)in[i + j];
  *(v8h*)(out + i) = r;
}

// ---------------------------------------------------------------------------
// Kernel 1: Q/K projection  Out = In @ W^T + bias (then scale), f16 in/out.
// Wave tile 32x64: 2 A-frags x 4 B-frags -> 8 WMMAs per 32-wide k-step.
// All fragment loads batched ahead of the WMMA burst for clause grouping.
// Output written head-major f16 [B,H,S,DK].
// ---------------------------------------------------------------------------
__global__ __launch_bounds__(256)
void proj_qk_kernel(const _Float16* __restrict__ In,  // [8192][1024]
                    const _Float16* __restrict__ W,   // [1024][1024]
                    const float* __restrict__ bias, _Float16* __restrict__ Out,
                    float scale) {
  const int tid  = threadIdx.x;
  const int wave = tid >> 5, lane = tid & 31;
  const int half = lane >> 4, l16 = lane & 15;
  const int g  = blockIdx.x * 8 + wave;          // 4096 wave tiles
  const int tm = g >> 4, tn = g & 15;            // 256 x 16
  const int m0 = tm * 32, n0 = tn * 64;

  const _Float16* a0 = In + (size_t)(m0 + l16) * NO_;
  const _Float16* a1 = In + (size_t)(m0 + 16 + l16) * NO_;
  const _Float16* br[4];
#pragma unroll
  for (int nt = 0; nt < 4; ++nt)
    br[nt] = W + (size_t)(n0 + nt * 16 + l16) * NO_;

  v8f acc[2][4] = {};
#pragma unroll 2
  for (int k0 = 0; k0 < NO_; k0 += 32) {
    if (k0 + 32 < NO_) __builtin_prefetch(a0 + k0 + 32, 0, 1);
    // batch all loads first so they land in one clause
    v16h A0 = afrag(a0, k0, half);
    v16h A1 = afrag(a1, k0, half);
    v16h Bf[4];
#pragma unroll
    for (int nt = 0; nt < 4; ++nt) Bf[nt] = bfrag(br[nt], k0, half);
#pragma unroll
    for (int nt = 0; nt < 4; ++nt) {
      acc[0][nt] = wmma16x16x32(A0, Bf[nt], acc[0][nt]);
      acc[1][nt] = wmma16x16x32(A1, Bf[nt], acc[1][nt]);
    }
  }

#pragma unroll
  for (int nt = 0; nt < 4; ++nt) {
    const int n = n0 + nt * 16 + l16;
    const int head = n >> 6, d = n & 63;
    const float bn = bias[n];
#pragma unroll
    for (int mr = 0; mr < 2; ++mr) {
#pragma unroll
      for (int j = 0; j < 8; ++j) {
        const int row = m0 + mr * 16 + j + half * 8;   // 0..8191
        const int b_ = row >> 11, s = row & (S_ - 1);
        Out[(((size_t)(b_ * H_ + head)) * S_ + s) * DK_ + d] =
            (_Float16)((acc[mr][nt][j] + bn) * scale);
      }
    }
  }
}

// ---------------------------------------------------------------------------
// Kernel 2: V projection per head: Out[.,e] = values[., h*64+d] @ Wv^T + bv
// Wave tile 16x64 (whole e-range of one head): 4 WMMAs per k-step.
// ---------------------------------------------------------------------------
__global__ __launch_bounds__(256)
void proj_v_kernel(const _Float16* __restrict__ V,   // [8192][1024] f16
                   const _Float16* __restrict__ Wv,  // [64][64] f16
                   const float* __restrict__ bv, _Float16* __restrict__ Out) {
  const int tid  = threadIdx.x;
  const int wave = tid >> 5, lane = tid & 31;
  const int half = lane >> 4, l16 = lane & 15;
  const int g  = blockIdx.x * 8 + wave;          // 8192 wave tiles
  const int tm = g >> 4, h = g & 15;             // 512 x 16 heads
  const int m0 = tm * 16;

  const _Float16* arow = V + (size_t)(m0 + l16) * NO_ + h * DK_;
  v8f acc[4] = {};
#pragma unroll
  for (int k0 = 0; k0 < DK_; k0 += 32) {
    v16h A = afrag(arow, k0, half);
    v16h Bf[4];
#pragma unroll
    for (int nt = 0; nt < 4; ++nt)
      Bf[nt] = bfrag(Wv + (size_t)(nt * 16 + l16) * DK_, k0, half);
#pragma unroll
    for (int nt = 0; nt < 4; ++nt) acc[nt] = wmma16x16x32(A, Bf[nt], acc[nt]);
  }
#pragma unroll
  for (int nt = 0; nt < 4; ++nt) {
    const int e = nt * 16 + l16;
    const float be = bv[e];
#pragma unroll
    for (int j = 0; j < 8; ++j) {
      const int row = m0 + j + half * 8;
      const int b_ = row >> 11, s = row & (S_ - 1);
      Out[(((size_t)(b_ * H_ + h)) * S_ + s) * DK_ + e] =
          (_Float16)(acc[nt][j] + be);
    }
  }
}

// ---------------------------------------------------------------------------
// Kernel 3: flash attention + head-sum.
// Block = 256 threads = 8 waves. Block owns (b, 16 query rows); each wave
// processes heads {wave, wave+8}. KV tile = 32 keys, online softmax.
// K tiles are staged global->LDS with async DMA (double-buffered, retired
// with s_wait_asynccnt); V is staged transposed in LDS; P is relaid out
// C->A through LDS.  Heads summed via LDS; output f16.
// ---------------------------------------------------------------------------
__global__ __launch_bounds__(256)
void attn_kernel(const _Float16* __restrict__ Qh, const _Float16* __restrict__ Kh,
                 const _Float16* __restrict__ Vh, _Float16* __restrict__ Xh) {
  __shared__ __align__(16) _Float16 Kl[8][2][32][DK_]; // 64 KB, K double buffer
  __shared__ __align__(16) _Float16 VT[8][DK_][32];    // 32 KB, V^T tile
  __shared__ __align__(16) _Float16 Pl[8][16][32];     //  8 KB, P tile
  __shared__ __align__(16) float    Red[8][16][DK_];   // 32 KB, head reduction

  const int tid  = threadIdx.x;
  const int wave = tid >> 5, lane = tid & 31;
  const int half = lane >> 4, l16 = lane & 15;
  const int b  = blockIdx.y;
  const int s0 = blockIdx.x * 16;

  v8f osum[4] = {};

  for (int hi = 0; hi < 2; ++hi) {
    const int h = wave + hi * 8;
    const size_t base = ((size_t)(b * H_ + h)) * S_ * DK_;

    // Q fragments (softmax scale folded in at projection time)
    const _Float16* qrow = Qh + base + (size_t)(s0 + l16) * DK_;
    v16h qa0 = afrag(qrow, 0, half);
    v16h qa1 = afrag(qrow, 32, half);

    float mrow[8], lrow[8];
#pragma unroll
    for (int j = 0; j < 8; ++j) { mrow[j] = -1e30f; lrow[j] = 0.f; }
    v8f acc[4] = {};

    // prologue: stage K tile 0 into buffer 0
    stage_tile_4k(Kh + base, &Kl[wave][0][0][0], lane);

    for (int it = 0; it < S_ / 32; ++it) {
      const int kb = it * 32;
      const int buf = it & 1;
      __syncthreads();  // prior iteration's VT/Pl reads are done
      // stage V tile transposed: VT[e][r], one key row per lane
      {
        const _Float16* vrow = Vh + base + (size_t)(kb + lane) * DK_;
#pragma unroll
        for (int eb = 0; eb < 8; ++eb) {
          v8h vv = ldg8h(vrow + eb * 8);
#pragma unroll
          for (int j = 0; j < 8; ++j) VT[wave][eb * 8 + j][lane] = vv[j];
        }
      }
      // prefetch next K tile into the other buffer, retire the current one
      if (kb + 32 < S_) {
        stage_tile_4k(Kh + base + (size_t)(kb + 32) * DK_,
                      &Kl[wave][buf ^ 1][0][0], lane);
        wait_async_le8();
      } else {
        wait_async_le0();
      }

      // scores: two 16x16 blocks over keys [kb, kb+32), K frags from LDS
      v8f sfr[2];
#pragma unroll
      for (int nt = 0; nt < 2; ++nt) {
        const _Float16* krow = &Kl[wave][buf][nt * 16 + l16][0];
        v8f s = {};
        s = wmma16x16x32(qa0, bfrag(krow, 0, half), s);
        s = wmma16x16x32(qa1, bfrag(krow, 32, half), s);
        sfr[nt] = s;
      }

      // online softmax update (rows split: half0 = rows j, half1 = rows j+8)
      float p0[8], p1[8];
#pragma unroll
      for (int j = 0; j < 8; ++j) {
        float t = fmaxf(sfr[0][j], sfr[1][j]);
#pragma unroll
        for (int m = 1; m < 16; m <<= 1) t = fmaxf(t, __shfl_xor(t, m, 32));
        const float nm = fmaxf(mrow[j], t);
        const float alpha = __expf(mrow[j] - nm);
        mrow[j] = nm;
        p0[j] = __expf(sfr[0][j] - nm);
        p1[j] = __expf(sfr[1][j] - nm);
        float rs = p0[j] + p1[j];
#pragma unroll
        for (int m = 1; m < 16; m <<= 1) rs += __shfl_xor(rs, m, 32);
        lrow[j] = lrow[j] * alpha + rs;
#pragma unroll
        for (int t4 = 0; t4 < 4; ++t4) acc[t4][j] = acc[t4][j] * alpha;
      }

      // stage P (16x32, C-layout) into LDS for A-fragment relayout
#pragma unroll
      for (int j = 0; j < 8; ++j) {
        Pl[wave][j + half * 8][l16]      = (_Float16)p0[j];
        Pl[wave][j + half * 8][16 + l16] = (_Float16)p1[j];
      }
      __syncthreads();

      const _Float16* prow = &Pl[wave][l16][0];
      v16h pa = afrag(prow, 0, half);
#pragma unroll
      for (int t4 = 0; t4 < 4; ++t4) {
        const _Float16* vtr = &VT[wave][t4 * 16 + l16][0];
        v16h vb = cat16(ldg8h(vtr + half * 8), ldg8h(vtr + 16 + half * 8));
        acc[t4] = wmma16x16x32(pa, vb, acc[t4]);
      }
    }  // it

    // normalize this head and accumulate into the per-wave head sum
#pragma unroll
    for (int j = 0; j < 8; ++j) {
      const float inv = 1.0f / lrow[j];
#pragma unroll
      for (int t4 = 0; t4 < 4; ++t4) osum[t4][j] += acc[t4][j] * inv;
    }
  }  // heads

  // cross-wave head reduction -> Xh[b, s0.., 0..63] (f16)
  __syncthreads();
#pragma unroll
  for (int t4 = 0; t4 < 4; ++t4)
#pragma unroll
    for (int j = 0; j < 8; ++j)
      Red[wave][j + half * 8][t4 * 16 + l16] = osum[t4][j];
  __syncthreads();

  for (int idx = tid; idx < 16 * DK_; idx += 256) {
    const int r = idx >> 6, c = idx & 63;
    float v = 0.f;
#pragma unroll
    for (int w = 0; w < 8; ++w) v += Red[w][r][c];
    Xh[(size_t)(b * S_ + s0 + r) * DK_ + c] = (_Float16)v;
  }
}

// ---------------------------------------------------------------------------
// Kernel 4: out = Xh @ Wo^T + bo.  Xh:[8192][64] f16, Woh:[1024][64] f16.
// Wave tile 16x64: 4 WMMAs per k-step, f32 output.
// ---------------------------------------------------------------------------
__global__ __launch_bounds__(256)
void out_proj_kernel(const _Float16* __restrict__ Xh,
                     const _Float16* __restrict__ Woh,
                     const float* __restrict__ bo, float* __restrict__ Out) {
  const int tid  = threadIdx.x;
  const int wave = tid >> 5, lane = tid & 31;
  const int half = lane >> 4, l16 = lane & 15;
  const int g  = blockIdx.x * 8 + wave;          // 8192 wave tiles
  const int tm = g >> 4, tn = g & 15;            // 512 x 16
  const int m0 = tm * 16, n0 = tn * 64;

  const _Float16* arow = Xh + (size_t)(m0 + l16) * DK_;
  v8f acc[4] = {};
#pragma unroll
  for (int k0 = 0; k0 < DK_; k0 += 32) {
    v16h A = afrag(arow, k0, half);
    v16h Bf[4];
#pragma unroll
    for (int nt = 0; nt < 4; ++nt)
      Bf[nt] = bfrag(Woh + (size_t)(n0 + nt * 16 + l16) * DK_, k0, half);
#pragma unroll
    for (int nt = 0; nt < 4; ++nt) acc[nt] = wmma16x16x32(A, Bf[nt], acc[nt]);
  }
#pragma unroll
  for (int nt = 0; nt < 4; ++nt) {
    const int n = n0 + nt * 16 + l16;
    const float bn = bo[n];
#pragma unroll
    for (int j = 0; j < 8; ++j) {
      const int row = m0 + j + half * 8;
      Out[(size_t)row * NO_ + n] = acc[nt][j] + bn;
    }
  }
}

// ---------------------------------------------------------------------------
extern "C" void kernel_launch(void* const* d_in, const int* in_sizes, int n_in,
                              void* d_out, int out_size, void* d_ws,
                              size_t ws_size, hipStream_t stream) {
  (void)in_sizes; (void)n_in; (void)out_size; (void)ws_size;
  const float* queries = (const float*)d_in[0];
  const float* keys    = (const float*)d_in[1];
  const float* values  = (const float*)d_in[2];
  const float* Wq = (const float*)d_in[3];
  const float* bq = (const float*)d_in[4];
  const float* Wk = (const float*)d_in[5];
  const float* bk = (const float*)d_in[6];
  const float* Wv = (const float*)d_in[7];
  const float* bv = (const float*)d_in[8];
  const float* Wo = (const float*)d_in[9];
  const float* bo = (const float*)d_in[10];

  const size_t tokh = (size_t)B_ * S_ * NO_;  // 8,388,608 == B*H*S*DK
  _Float16* p  = (_Float16*)d_ws;
  _Float16* Qf  = p; p += tokh;               // f16 copies of activations
  _Float16* Kf  = p; p += tokh;
  _Float16* Vf  = p; p += tokh;
  _Float16* Wqh = p; p += (size_t)NO_ * NO_;  // f16 copies of weights
  _Float16* Wkh = p; p += (size_t)NO_ * NO_;
  _Float16* Wvh = p; p += (size_t)DK_ * DK_;
  _Float16* Woh = p; p += (size_t)NO_ * DK_;
  _Float16* Qh  = p; p += tokh;               // projected, head-major
  _Float16* Kh  = p; p += tokh;
  _Float16* Vh  = p; p += tokh;
  _Float16* Xh  = p; p += (size_t)B_ * S_ * DK_;  // head-summed attention out

  dim3 blk(256);
  // fp32 -> f16 conversion passes (2048 elements per block)
  cvt_f16_kernel<<<tokh / 2048, blk, 0, stream>>>(queries, Qf);
  cvt_f16_kernel<<<tokh / 2048, blk, 0, stream>>>(keys, Kf);
  cvt_f16_kernel<<<tokh / 2048, blk, 0, stream>>>(values, Vf);
  cvt_f16_kernel<<<(NO_ * NO_) / 2048, blk, 0, stream>>>(Wq, Wqh);
  cvt_f16_kernel<<<(NO_ * NO_) / 2048, blk, 0, stream>>>(Wk, Wkh);
  cvt_f16_kernel<<<(DK_ * DK_) / 2048, blk, 0, stream>>>(Wv, Wvh);
  cvt_f16_kernel<<<(NO_ * DK_) / 2048, blk, 0, stream>>>(Wo, Woh);

  // 1/sqrt(d_k) = 0.125 folded into Q at projection time
  proj_qk_kernel<<<512, blk, 0, stream>>>(Qf, Wqh, bq, Qh, 0.125f);
  proj_qk_kernel<<<512, blk, 0, stream>>>(Kf, Wkh, bk, Kh, 1.0f);
  proj_v_kernel<<<1024, blk, 0, stream>>>(Vf, Wvh, bv, Vh);
  attn_kernel<<<dim3(S_ / 16, B_), blk, 0, stream>>>(Qh, Kh, Vh, Xh);
  out_proj_kernel<<<1024, blk, 0, stream>>>(Xh, Woh, bo, (float*)d_out);
}
